// RecurrentForgettingAttention_74775380623499
// MI455X (gfx1250) — matrix operand order repaired and checked by reading, not compiled, not run on hardware
//
#include <hip/hip_runtime.h>
#include <hip/hip_bf16.h>

// ---------------------------------------------------------------------------
// RecurrentForgettingAttention (B=8,S=512,D=256) for MI455X / gfx1250
//
// Chunkwise-parallel reformulation of the scalar-gated linear-attention scan:
//   P_i   = prod_{j<=i} f_j                         (per chunk, f in (0,1))
//   num_i = P_i q_i @ S_in + sum_{m<=i} (P_i/P_m)(q_i.k_m) v_m
//   den_i = P_i (q_i.z_in) + rowsum_i(maskedA) + eps
//   S_out = Pend S_in + sum_m (Pend/P_m) k_m v_m^T ; z_out analogous.
// All GEMMs run on v_wmma_f32_16x16x32_bf16 (f32 accumulate). The recurrent
// state is accumulated in f32 (St) with a bf16 shadow (StB) maintained in the
// same writeback pass (packed b128 stores), so GEMM1's B-operand is a plain
// 2x ds_load_b128 fragment instead of 16 scalar f32 loads + 8 converts.
// Chunk Q/K tiles are staged global->LDS with GLOBAL_LOAD_ASYNC_TO_LDS_B128.
// ---------------------------------------------------------------------------

typedef __bf16 bf16_t;
typedef __attribute__((ext_vector_type(16))) __bf16 v16bf;
typedef __attribute__((ext_vector_type(8)))  __bf16 v8bf;
typedef __attribute__((ext_vector_type(8)))  float  v8f;
typedef __attribute__((ext_vector_type(4)))  float  v4f;

constexpr int BB = 8;
constexpr int SS = 512;
constexpr int DD = 256;
constexpr int MM = BB * SS;    // 4096
constexpr int CH = 64;         // chunk length
constexpr int DV = 64;         // dv tile per workgroup
constexpr float EPSV = 1e-6f;

// ---------------------------------------------------------------------------
// WMMA helpers (layouts per CDNA5 ISA 7.12.2, wave32)
// ---------------------------------------------------------------------------
__device__ __forceinline__ v8f wmma_bf16(v16bf a, v16bf b, v8f c) {
  return __builtin_amdgcn_wmma_f32_16x16x32_bf16(
      /*neg_a=*/false, a, /*neg_b=*/false, b, /*c_mod=*/(short)0, c,
      /*reuse_a=*/false, /*reuse_b=*/false);
}

// A fragment: 16x32 (MxK), source row-major, ld elements. Lane l (both
// halves) holds row M=l&15; dwords 0..3 = K(hl*8 .. +7), 4..7 = K(16+hl*8..).
__device__ __forceinline__ v16bf frag_a_bf16(const bf16_t* tile, int ld, int lane) {
  const int hl = lane >> 4, r = lane & 15;
  const bf16_t* p = tile + r * ld + hl * 8;
  v8bf lo = *(const v8bf*)(p);
  v8bf hi = *(const v8bf*)(p + 16);
  v16bf f;
#pragma unroll
  for (int i = 0; i < 8; ++i) { f[i] = lo[i]; f[i + 8] = hi[i]; }
  return f;
}

// B fragment: 32x16 (KxN), source stored TRANSPOSED (N x K row-major) so each
// lane's K-values are contiguous runs. dwords 0..3 = K(hl*8 .. +7), 4..7 = K(16+hl*8..).
__device__ __forceinline__ v16bf frag_b_bf16(const bf16_t* tileT, int ld, int lane) {
  const int hl = lane >> 4, r = lane & 15;
  const bf16_t* p = tileT + r * ld + hl * 8;
  v8bf lo = *(const v8bf*)(p);
  v8bf hi = *(const v8bf*)(p + 16);
  v16bf f;
#pragma unroll
  for (int i = 0; i < 8; ++i) { f[i] = lo[i]; f[i + 8] = hi[i]; }
  return f;
}

// ---------------------------------------------------------------------------
// gfx1250 async global->LDS copy (ASYNCcnt-tracked). LDS operand is the
// wave-relative LDS byte offset (HW adds LDS_BASE). CDNA5 ISA.
// ---------------------------------------------------------------------------
__device__ __forceinline__ void async_ld_b128(unsigned lds_off, const void* gaddr) {
  asm volatile("global_load_async_to_lds_b128 %0, %1, off"
               :: "v"(lds_off), "v"(gaddr)
               : "memory");
}
__device__ __forceinline__ void wait_asynccnt0() {
  asm volatile("s_wait_asynccnt 0x0" ::: "memory");
}

// ---------------------------------------------------------------------------
// Prep kernels
// ---------------------------------------------------------------------------
__global__ void __launch_bounds__(256) k_convert_bf16(const float* __restrict__ in,
                                                      bf16_t* __restrict__ out, int n) {
  int i = blockIdx.x * 256 + threadIdx.x;
  if (i < n) out[i] = (bf16_t)in[i];
}

// Wt[n][k] = W[k][n], f32 -> bf16 (so W becomes an N x K "Bt" operand)
__global__ void __launch_bounds__(256) k_transpose_bf16(const float* __restrict__ W,
                                                        bf16_t* __restrict__ Wt, int d) {
  int i = blockIdx.x * 256 + threadIdx.x;
  if (i < d * d) {
    int n = i / d, k = i - n * d;
    Wt[n * d + k] = (bf16_t)W[k * d + n];
  }
}

// ---------------------------------------------------------------------------
// Generic WMMA GEMM: C[MxN] = A(bf16 MxK rowmajor) * Bt(bf16 NxK rowmajor)^T + bias
// N=K=256 compile-time. One wave owns one 16-row M tile; the whole A row-strip
// (16x256 = 8 frags) is register-cached, then B-frags stream from L2-resident
// global (whole problem ~15MB vs 192MB L2).
// ---------------------------------------------------------------------------
template <int OUT_F32>
__global__ void __launch_bounds__(256) k_gemm(const bf16_t* __restrict__ A,
                                              const bf16_t* __restrict__ Bt,
                                              const float* __restrict__ bias,
                                              void* __restrict__ Cout, int M) {
  constexpr int N = DD, K = DD;
  const int lane = threadIdx.x & 31;
  const int wave = threadIdx.x >> 5;
  const int mtile = blockIdx.x * 8 + wave;
  if (mtile * 16 >= M) return;  // wave-uniform; EXEC stays all-1 for WMMA
  const bf16_t* Arow = A + (size_t)mtile * 16 * K;
  v16bf areg[8];
#pragma unroll
  for (int kk = 0; kk < 8; ++kk) areg[kk] = frag_a_bf16(Arow + kk * 32, K, lane);
  for (int nt = 0; nt < (N >> 4); ++nt) {
    v8f acc = {};
    const bf16_t* Btt = Bt + (size_t)nt * 16 * K;
#pragma unroll
    for (int kk = 0; kk < 8; ++kk) {
      v16bf bfr = frag_b_bf16(Btt + kk * 32, K, lane);
      acc = wmma_bf16(areg[kk], bfr, acc);
    }
    const int col = (nt << 4) + (lane & 15);
    const float bvv = bias[col];
#pragma unroll
    for (int r = 0; r < 8; ++r) {
      const int row = mtile * 16 + r + ((lane >> 4) << 3);
      const float v = acc[r] + bvv;
      if (OUT_F32) ((float*)Cout)[(size_t)row * N + col] = v;
      else         ((bf16_t*)Cout)[(size_t)row * N + col] = (bf16_t)v;
    }
  }
}

// ---------------------------------------------------------------------------
// Chunked recurrent scan. Grid: B*4 workgroups (batch x dv-tile of 64),
// 256 threads = 8 waves. State tile (256k x 64dv) f32 + bf16 shadow live in
// LDS across the 8 sequential chunks; ~253KB dynamic LDS (< 320KB/WGP).
// ---------------------------------------------------------------------------
__global__ void __launch_bounds__(256) k_rfa_scan(const bf16_t* __restrict__ Qb,
                                                  const bf16_t* __restrict__ Kb,
                                                  const bf16_t* __restrict__ Vb,
                                                  const float* __restrict__ gwq,
                                                  const float* __restrict__ gwk,
                                                  const float* __restrict__ gbp,
                                                  bf16_t* __restrict__ attn) {
  constexpr int LD_ST = 260;   // f32 state stride (260%64=4 -> conflict-free)
  constexpr int LD_SB = 264;   // bf16 shadow stride (16B-aligned rows)
  constexpr int LD_T  = 72;    // bf16 small-tile stride (16B-aligned rows)
  extern __shared__ char smem[];
  float*  St   = (float*)(smem);                 // [DV][260] f32   66560 B  state^T [dv][k]
  bf16_t* StB  = (bf16_t*)(smem + 66560);        // [DV][264] bf16  33792 B  bf16 shadow
  bf16_t* Qc   = (bf16_t*)(smem + 100352);       // [64][256] bf16  32768 B
  bf16_t* Kc   = (bf16_t*)(smem + 133120);       // [64][256] bf16  32768 B
  bf16_t* Qp   = (bf16_t*)(smem + 165888);       // [64][256] bf16  32768 B  P_t * q_t
  bf16_t* Vt   = (bf16_t*)(smem + 198656);       // [64dv][72] bf16  9216 B  V^T
  bf16_t* KppT = (bf16_t*)(smem + 207872);       // [256k][72] bf16 36864 B  decayed K^T
  bf16_t* Am   = (bf16_t*)(smem + 244736);       // [64][72] bf16    9216 B  masked attn
  float*  zv   = (float*)(smem + 253952);        // [256] z state
  float*  red  = (float*)(smem + 254976);        // [256] reduce scratch
  float*  Pv   = (float*)(smem + 256000);        // [64] cumulative decay
  float*  fv   = (float*)(smem + 256256);        // [64] gates, then 1/P_t
  float*  qz   = (float*)(smem + 256512);        // [64] q.z_in
  float*  denv = (float*)(smem + 256768);        // [64] 1/denominator
  float*  g1   = (float*)(smem + 257024);        // [256] g_wq
  float*  g2   = (float*)(smem + 258048);        // [256] g_wk
  // total 259072 bytes

  const int tid = threadIdx.x, lane = tid & 31, wave = tid >> 5;
  const int b   = blockIdx.x >> 2;
  const int dv0 = (blockIdx.x & 3) * DV;

  for (int i = tid; i < DD; i += 256) { g1[i] = gwq[i]; g2[i] = gwk[i]; zv[i] = 0.f; }
  for (int i = tid; i < DV * LD_ST; i += 256) St[i] = 0.f;
  for (int i = tid; i < DV * LD_SB; i += 256) StB[i] = (bf16_t)0.f;
  const float gbias = gbp[0];
  __syncthreads();

  const bf16_t* Qg = Qb + (size_t)b * SS * DD;
  const bf16_t* Kg = Kb + (size_t)b * SS * DD;
  const bf16_t* Vg = Vb + (size_t)b * SS * DD;
  bf16_t*       Og = attn + (size_t)b * SS * DD;

  const unsigned qc_off = (unsigned)((const char*)Qc - smem);
  const unsigned kc_off = (unsigned)((const char*)Kc - smem);

  for (int chI = 0; chI < SS / CH; ++chI) {
    const int t0 = chI * CH;
    // ---- chunk loads: Q,K async global->LDS (b128); V transposed to Vt ----
    {
      const bf16_t* qsrc = Qg + (size_t)t0 * DD;
      const bf16_t* ksrc = Kg + (size_t)t0 * DD;
      for (int i = tid; i < CH * DD / 8; i += 256) {
        async_ld_b128(qc_off + (unsigned)i * 16u, qsrc + i * 8);
        async_ld_b128(kc_off + (unsigned)i * 16u, ksrc + i * 8);
      }
      for (int i = tid; i < CH * DV; i += 256) {
        const int t = i >> 6, j = i & 63;
        Vt[j * LD_T + t] = Vg[(size_t)(t0 + t) * DD + dv0 + j];
      }
      wait_asynccnt0();
    }
    __syncthreads();

    // ---- gates: f_t = sigmoid(q_t.gwq + k_t.gwk + b)  (vector LDS reads) ----
    {
      const int t = tid >> 2, q4 = tid & 3;
      const v8bf* qrow = (const v8bf*)(Qc + t * DD + q4 * 64);
      const v8bf* krow = (const v8bf*)(Kc + t * DD + q4 * 64);
      float s = 0.f;
#pragma unroll
      for (int c8 = 0; c8 < 8; ++c8) {
        v8bf qv = qrow[c8], kv = krow[c8];
        const float* gq = g1 + q4 * 64 + c8 * 8;
        const float* gk = g2 + q4 * 64 + c8 * 8;
#pragma unroll
        for (int e = 0; e < 8; ++e) s += (float)qv[e] * gq[e] + (float)kv[e] * gk[e];
      }
      red[tid] = s;
    }
    __syncthreads();
    if (tid < CH) {
      const float x = red[tid * 4] + red[tid * 4 + 1] + red[tid * 4 + 2] + red[tid * 4 + 3] + gbias;
      fv[tid] = 1.f / (1.f + __expf(-x));
    }
    __syncthreads();
    if (tid == 0) { float p = 1.f; for (int t = 0; t < CH; ++t) { p *= fv[t]; Pv[t] = p; } }
    __syncthreads();
    // ---- qz = q_t . z_in ; Pinv = 1/P_t (fv reused) ----
    {
      const int t = tid >> 2, q4 = tid & 3;
      const v8bf* qrow = (const v8bf*)(Qc + t * DD + q4 * 64);
      float s = 0.f;
#pragma unroll
      for (int c8 = 0; c8 < 8; ++c8) {
        v8bf qv = qrow[c8];
        const float* zp = zv + q4 * 64 + c8 * 8;
#pragma unroll
        for (int e = 0; e < 8; ++e) s += (float)qv[e] * zp[e];
      }
      red[tid] = s;
    }
    __syncthreads();
    float* Pinv = fv;  // gates consumed; reuse as reciprocal decay
    if (tid < CH) {
      qz[tid] = red[tid * 4] + red[tid * 4 + 1] + red[tid * 4 + 2] + red[tid * 4 + 3];
      Pinv[tid] = 1.f / Pv[tid];
    }
    __syncthreads();

    // ---- GEMM2: Am[i][m] = (q_i.k_m)*P_i/P_m, causal-masked.
    //      Tile classes (wave-uniform): upper -> zeros, diag -> select,
    //      lower -> no mask. Branchless per element (no exec toggling). ----
    for (int tt = 0; tt < 2; ++tt) {
      const int tileid = wave * 2 + tt;
      const int mi = (tileid >> 2) << 4;
      const int ni = (tileid & 3) << 4;
      const int mcol = ni + (lane & 15);
      const int rbase = mi + ((lane >> 4) << 3);
      if (ni > mi) {                       // strictly upper: all zero
#pragma unroll
        for (int r = 0; r < 8; ++r) Am[(rbase + r) * LD_T + mcol] = (bf16_t)0.f;
      } else {
        v8f acc = {};
        for (int kk = 0; kk < DD; kk += 32) {
          v16bf af = frag_a_bf16(Qc + mi * DD + kk, DD, lane);
          v16bf bf = frag_b_bf16(Kc + ni * DD + kk, DD, lane);
          acc = wmma_bf16(af, bf, acc);
        }
        const float pim = Pinv[mcol];
        float pvr[8];
#pragma unroll
        for (int r = 0; r < 8; ++r) pvr[r] = Pv[rbase + r];
        if (ni == mi) {                    // diagonal: branchless select
#pragma unroll
          for (int r = 0; r < 8; ++r) {
            const int irow = rbase + r;
            float v = acc[r] * (pvr[r] * pim);
            v = (mcol <= irow) ? v : 0.f;
            Am[irow * LD_T + mcol] = (bf16_t)v;
          }
        } else {                           // strictly lower: no mask needed
#pragma unroll
          for (int r = 0; r < 8; ++r) {
            const int irow = rbase + r;
            Am[irow * LD_T + mcol] = (bf16_t)(acc[r] * (pvr[r] * pim));
          }
        }
      }
    }
    __syncthreads();

    // ---- rowsum(Am); KppT[k][t]=(Pend/P_t)K[t][k]; Qp[t][k]=P_t*Q[t][k] ----
    {
      const int t = tid >> 2, q4 = tid & 3;
      float s = 0.f;
      for (int m = q4 * 16; m < q4 * 16 + 16; ++m) s += (float)Am[t * LD_T + m];
      red[tid] = s;
      const float Pend = Pv[CH - 1];
      for (int i = tid; i < CH * DD; i += 256) {
        const int tt2 = i & 63, k = i >> 6;
        KppT[k * LD_T + tt2] = (bf16_t)((float)Kc[tt2 * DD + k] * (Pend * Pinv[tt2]));
      }
      for (int i = tid; i < CH * DD; i += 256) {
        const int tq = i >> 8;
        Qp[i] = (bf16_t)((float)Qc[i] * Pv[tq]);
      }
    }
    __syncthreads();
    if (tid < CH)
      denv[tid] = 1.f / (Pv[tid] * qz[tid] +
                         red[tid * 4] + red[tid * 4 + 1] + red[tid * 4 + 2] + red[tid * 4 + 3] +
                         EPSV);
    __syncthreads();

    // ---- GEMM1 (Qp @ S_in, bf16 shadow) + GEMM3 (Am @ V) -> out, * (1/den).
    //      GEMM3 skips k-steps that only touch zeroed upper-triangle Am. ----
    for (int tt = 0; tt < 2; ++tt) {
      const int tileid = wave * 2 + tt;
      const int mi = (tileid >> 2) << 4;   // t rows
      const int nj = (tileid & 3) << 4;    // dv cols
      v8f acc = {};
      for (int kk = 0; kk < DD; kk += 32) {
        v16bf af = frag_a_bf16(Qp + mi * DD + kk, DD, lane);
        v16bf bf = frag_b_bf16(StB + nj * LD_SB + kk, LD_SB, lane);
        acc = wmma_bf16(af, bf, acc);
      }
      const int kmax = (mi >= 32) ? CH : 32;  // rows t<32: Am[t][m>=32]==0
      for (int kk = 0; kk < kmax; kk += 32) {
        v16bf af = frag_a_bf16(Am + mi * LD_T + kk, LD_T, lane);
        v16bf bf = frag_b_bf16(Vt + nj * LD_T + kk, LD_T, lane);
        acc = wmma_bf16(af, bf, acc);
      }
      const int col = dv0 + nj + (lane & 15);
      const int rbase = mi + ((lane >> 4) << 3);
      float dvr[8];
#pragma unroll
      for (int r = 0; r < 8; ++r) dvr[r] = denv[rbase + r];
#pragma unroll
      for (int r = 0; r < 8; ++r)
        Og[(size_t)(t0 + rbase + r) * DD + col] = (bf16_t)(acc[r] * dvr[r]);
    }
    __syncthreads();   // all S_in reads complete before state update

    // ---- GEMM4: St = Pend*St + KppT @ V  (64 tiles, 8/wave, disjoint).
    //      C-frag rows are 8 consecutive k -> vector init/writeback + packed
    //      bf16 shadow store (single ds_store_b128). ----
    {
      const float Pend = Pv[CH - 1];
      for (int u = 0; u < 8; ++u) {
        const int tileid = wave * 8 + u;
        const int mi = (tileid >> 2) << 4;  // k rows
        const int nj = (tileid & 3) << 4;   // dv cols
        const int col = nj + (lane & 15);
        const int kbase = mi + ((lane >> 4) << 3);
        float* srow = St + col * LD_ST + kbase;       // 8 consecutive f32, 16B-aligned
        v4f a0 = *(const v4f*)(srow);
        v4f a1 = *(const v4f*)(srow + 4);
        v8f acc;
#pragma unroll
        for (int r = 0; r < 4; ++r) { acc[r] = Pend * a0[r]; acc[r + 4] = Pend * a1[r]; }
        for (int kk = 0; kk < CH; kk += 32) {
          v16bf af = frag_a_bf16(KppT + mi * LD_T + kk, LD_T, lane);
          v16bf bf = frag_b_bf16(Vt + nj * LD_T + kk, LD_T, lane);
          acc = wmma_bf16(af, bf, acc);
        }
        v4f w0, w1;
        v8bf pb;
#pragma unroll
        for (int r = 0; r < 4; ++r) {
          w0[r] = acc[r]; w1[r] = acc[r + 4];
          pb[r] = (bf16_t)acc[r]; pb[r + 4] = (bf16_t)acc[r + 4];
        }
        *(v4f*)(srow) = w0;
        *(v4f*)(srow + 4) = w1;
        *(v8bf*)(StB + col * LD_SB + kbase) = pb;     // 16B packed shadow store
      }
      // z update: z = Pend*z + sum_m (Pend/P_m) k_m   (thread-per-k, no races)
      {
        float a = Pend * zv[tid];
        for (int m = 0; m < CH; ++m) a += (Pend * Pinv[m]) * (float)Kc[m * DD + tid];
        zv[tid] = a;
      }
    }
    __syncthreads();
  }
}

// ---------------------------------------------------------------------------
// Host launcher
// ---------------------------------------------------------------------------
extern "C" void kernel_launch(void* const* d_in, const int* in_sizes, int n_in,
                              void* d_out, int out_size, void* d_ws, size_t ws_size,
                              hipStream_t stream) {
  const float* x   = (const float*)d_in[0];
  const float* Wq  = (const float*)d_in[1];
  const float* bq  = (const float*)d_in[2];
  const float* Wk  = (const float*)d_in[3];
  const float* bk  = (const float*)d_in[4];
  const float* Wv  = (const float*)d_in[5];
  const float* bv  = (const float*)d_in[6];
  const float* Wo  = (const float*)d_in[7];
  const float* bo  = (const float*)d_in[8];
  const float* gwq = (const float*)d_in[9];
  const float* gwk = (const float*)d_in[10];
  const float* gb  = (const float*)d_in[11];
  float* out = (float*)d_out;

  char* ws = (char*)d_ws;
  size_t o = 0;
  bf16_t* xb  = (bf16_t*)(ws + o); o += (size_t)MM * DD * 2;
  bf16_t* Wtq = (bf16_t*)(ws + o); o += (size_t)DD * DD * 2;
  bf16_t* Wtk = (bf16_t*)(ws + o); o += (size_t)DD * DD * 2;
  bf16_t* Wtv = (bf16_t*)(ws + o); o += (size_t)DD * DD * 2;
  bf16_t* Wto = (bf16_t*)(ws + o); o += (size_t)DD * DD * 2;
  bf16_t* Qb  = (bf16_t*)(ws + o); o += (size_t)MM * DD * 2;
  bf16_t* Kb  = (bf16_t*)(ws + o); o += (size_t)MM * DD * 2;
  bf16_t* Vb  = (bf16_t*)(ws + o); o += (size_t)MM * DD * 2;
  bf16_t* At  = (bf16_t*)(ws + o); o += (size_t)MM * DD * 2;   // ~10.4 MB total

  // prep: bf16 activations + transposed bf16 weights (N x K "Bt" layout)
  k_convert_bf16<<<(MM * DD) / 256, 256, 0, stream>>>(x, xb, MM * DD);
  k_transpose_bf16<<<(DD * DD) / 256, 256, 0, stream>>>(Wq, Wtq, DD);
  k_transpose_bf16<<<(DD * DD) / 256, 256, 0, stream>>>(Wk, Wtk, DD);
  k_transpose_bf16<<<(DD * DD) / 256, 256, 0, stream>>>(Wv, Wtv, DD);
  k_transpose_bf16<<<(DD * DD) / 256, 256, 0, stream>>>(Wo, Wto, DD);

  // QKV projections (bf16 out)
  k_gemm<0><<<MM / 128, 256, 0, stream>>>(xb, Wtq, bq, Qb, MM);
  k_gemm<0><<<MM / 128, 256, 0, stream>>>(xb, Wtk, bk, Kb, MM);
  k_gemm<0><<<MM / 128, 256, 0, stream>>>(xb, Wtv, bv, Vb, MM);

  // chunked recurrent scan (B*4 workgroups, 253KB dynamic LDS each)
  k_rfa_scan<<<BB * 4, 256, 259072, stream>>>(Qb, Kb, Vb, gwq, gwk, gb, At);

  // output projection (f32 out + bias)
  k_gemm<1><<<MM / 128, 256, 0, stream>>>(At, Wto, bo, out, MM);
}
